// RoIHeads_69887707840901
// MI455X (gfx1250) — compile-verified, matrix operand order checked by reading, not run on hardware
//
#include <hip/hip_runtime.h>
#include <math.h>

// ---------------- problem constants (match reference) ----------------
#define NPROP      50000
#define NCLS       91          // classes incl. background
#define NC         90          // foreground classes
#define NT         3           // trajectory steps
#define NUM_DET    100
#define M_OUT      (NC * NUM_DET)      // 9000
#define SCORE_TH   0.05f
#define NMS_TH     0.5f
#define MIN_SIZE   1.0f
#define BBOX_CLIP  4.135166556742356f  // log(1000/16)
#define CAP_MAX    2048

typedef __attribute__((ext_vector_type(16))) _Float16 v16h;
typedef __attribute__((ext_vector_type(8)))  float    v8f;
typedef int v4i __attribute__((vector_size(16)));   // matches builtin param type

// gfx1250 async global->LDS staging (ASYNCcnt-tracked), guarded so the file
// compiles on toolchains without the builtins.
#if defined(__has_builtin)
# if __has_builtin(__builtin_amdgcn_global_load_async_to_lds_b32) && \
     __has_builtin(__builtin_amdgcn_global_load_async_to_lds_b128)
#  define USE_ASYNC_LDS 1
# endif
#endif
#ifndef USE_ASYNC_LDS
# define USE_ASYNC_LDS 0
#endif

#if USE_ASYNC_LDS
// AS3 pointers are 32-bit; generic LDS addresses carry the LDS byte offset in
// their low 32 bits (ISA 00_overview §10.2 aperture mapping), so truncate.
#define TO_LDS32(p)  ((__attribute__((address_space(3))) int*)(unsigned int)(unsigned long long)(p))
#define TO_GLB32(p)  ((__attribute__((address_space(1))) int*)(unsigned long long)(p))
#define TO_LDS128(p) ((__attribute__((address_space(3))) v4i*)(unsigned int)(unsigned long long)(p))
#define TO_GLB128(p) ((__attribute__((address_space(1))) v4i*)(unsigned long long)(p))
#endif

// workspace layout (bytes):
//   [0 .. 360)                 : int counters[90]
//   [512 + c*cap*24 .. )       : class c: score f32[cap] | box f32[4*cap] | idx i32[cap]
__device__ __forceinline__ float* ws_score(unsigned char* ws, int cap, int c) {
    return (float*)(ws + 512 + (size_t)c * cap * 24);
}
__device__ __forceinline__ float* ws_box(unsigned char* ws, int cap, int c) {
    return (float*)(ws + 512 + (size_t)c * cap * 24 + (size_t)cap * 4);
}
__device__ __forceinline__ int* ws_idx(unsigned char* ws, int cap, int c) {
    return (int*)(ws + 512 + (size_t)c * cap * 24 + (size_t)cap * 20);
}

__device__ __forceinline__ float clampf(float v, float lo, float hi) {
    return fminf(fmaxf(v, lo), hi);
}

// Faster-RCNN BoxCoder.decode, reg_weights (10,10,5,5)
__device__ __forceinline__ void decode1(float dx, float dy, float dw, float dh,
                                        float x1, float y1, float x2, float y2,
                                        float& ox1, float& oy1, float& ox2, float& oy2) {
    float w  = x2 - x1;
    float h  = y2 - y1;
    float cx = x1 + 0.5f * w;
    float cy = y1 + 0.5f * h;
    dx *= 0.1f;
    dy *= 0.1f;
    dw = fminf(dw * 0.2f, BBOX_CLIP);
    dh = fminf(dh * 0.2f, BBOX_CLIP);
    float pcx = dx * w + cx;
    float pcy = dy * h + cy;
    float pw  = expf(dw) * w;
    float ph  = expf(dh) * h;
    ox1 = pcx - 0.5f * pw;
    oy1 = pcy - 0.5f * ph;
    ox2 = pcx + 0.5f * pw;
    oy2 = pcy + 0.5f * ph;
}

// ---------------------------------------------------------------------
__global__ void k_zero_counters(int* counters) {
    if (threadIdx.x < NC) counters[threadIdx.x] = 0;
}

// ---------------------------------------------------------------------
// Kernel B: softmax denominators via WMMA row-reduction, then per-(row,class)
// filter + box decode + candidate append.
// 256 threads = 8 waves; each wave reduces 16 rows; block covers 128 rows.
__global__ void __launch_bounds__(256)
k_softmax_filter(const float* __restrict__ logits,
                 const float* __restrict__ boxreg,
                 const float* __restrict__ proposal,
                 unsigned char* __restrict__ ws, int cap,
                 const int* __restrict__ pH, const int* __restrict__ pW) {
    __shared__ float s_inv[128];

    const int tid  = threadIdx.x;
    const int wv   = tid >> 5;
    const int lane = tid & 31;
    const int g    = lane >> 4;        // half-wave group
    const int r    = lane & 15;        // row within tile
    const int blockRow0 = blockIdx.x * 128;
    const int row0 = blockRow0 + wv * 16;

    // ---- phase 1: rowsum(exp(logits)) for 16 rows via v_wmma_f32_16x16x32_f16
    {
        int row  = row0 + r;
        int lrow = row < NPROP ? row : (NPROP - 1);   // clamp so EXEC stays full
        const float* lrp = logits + (size_t)lrow * NCLS;

        v16h bones;
#pragma unroll
        for (int i = 0; i < 16; ++i) bones[i] = (_Float16)1.0f;

        v8f acc = {};
#pragma unroll
        for (int chunk = 0; chunk < 3; ++chunk) {
            v16h a;
#pragma unroll
            for (int v = 0; v < 8; ++v) {
                // 16-bit A 16x32 layout: VGPR v<4 -> K=g*8+2v ; v>=4 -> K=16+g*8+2(v-4)
                int k0 = chunk * 32 + ((v < 4) ? (g * 8 + 2 * v)
                                               : (16 + g * 8 + 2 * (v - 4)));
                // branchless tail: clamp the address, zero the value
                float l0 = lrp[(k0     < NCLS) ? k0     : 0];
                float l1 = lrp[(k0 + 1 < NCLS) ? k0 + 1 : 0];
                float e0 = (k0     < NCLS) ? __expf(l0) : 0.0f;
                float e1 = (k0 + 1 < NCLS) ? __expf(l1) : 0.0f;
                a[2 * v]     = (_Float16)e0;
                a[2 * v + 1] = (_Float16)e1;
            }
            acc = __builtin_amdgcn_wmma_f32_16x16x32_f16(
                false, a, false, bones, (short)0, acc, false, false);
        }
        // D layout: VGPR j, lanes[0..15] hold M=j, lanes[16..31] hold M=j+8.
        // All columns equal (B = ones) -> acc[j] = rowsum(row0 + g*8 + j).
#pragma unroll
        for (int j = 0; j < 8; ++j) {
            s_inv[wv * 16 + g * 8 + j] = 1.0f / acc[j];
        }
    }
    __syncthreads();

    // ---- phase 2: filter + decode + append candidates
    const float imw = (float)(*pW);
    const float imh = (float)(*pH);
    int* counters = (int*)ws;

    const int pairs = 128 * NC;
    for (int p = tid; p < pairs; p += 256) {
        int rowLocal = p / NC;
        int c        = p - rowLocal * NC;   // foreground class index 0..89
        int row      = blockRow0 + rowLocal;
        if (row >= NPROP) continue;

        float score = __expf(logits[(size_t)row * NCLS + (c + 1)]) * s_inv[rowLocal];
        if (!(score >= SCORE_TH)) continue;

        float4 pr = ((const float4*)proposal)[row];
        float4 dl = ((const float4*)boxreg)[(size_t)row * NCLS + (c + 1)];
        float x1, y1, x2, y2;
        decode1(dl.x, dl.y, dl.z, dl.w, pr.x, pr.y, pr.z, pr.w, x1, y1, x2, y2);
        x1 = clampf(x1, 0.0f, imw);
        y1 = clampf(y1, 0.0f, imh);
        x2 = clampf(x2, 0.0f, imw);
        y2 = clampf(y2, 0.0f, imh);
        if ((x2 - x1) < MIN_SIZE || (y2 - y1) < MIN_SIZE) continue;

        int pos = atomicAdd(&counters[c], 1);
        if (pos < cap) {
            ws_score(ws, cap, c)[pos] = score;
            float* bb = ws_box(ws, cap, c) + (size_t)pos * 4;
            bb[0] = x1; bb[1] = y1; bb[2] = x2; bb[3] = y2;
            ws_idx(ws, cap, c)[pos] = row;
        }
    }
}

// ---------------------------------------------------------------------
// Kernel C: per-class greedy NMS in LDS + outputs + trajectory decode.
__global__ void __launch_bounds__(256)
k_nms_output(unsigned char* __restrict__ ws, int cap,
             const float* __restrict__ traj,
             const float* __restrict__ proposal,
             float* __restrict__ out,
             const int* __restrict__ pH, const int* __restrict__ pW) {
    __shared__ float s_score[CAP_MAX];
    __shared__ __align__(16) float s_box[CAP_MAX * 4];
    __shared__ int   s_idx[CAP_MAX];
    __shared__ float red_val[256];
    __shared__ int   red_idx[256];
    __shared__ int   s_keep[NUM_DET];
    __shared__ float s_kscore[NUM_DET];
    __shared__ float s_bb[4];
    __shared__ int   s_bi;

    const int tid = threadIdx.x;
    const int c   = blockIdx.x;            // foreground class 0..89
    const float NEG = -__builtin_inff();

    int cnt = ((int*)ws)[c];
    if (cnt > cap) cnt = cap;

    // stage candidates into LDS
    const float* gsc = ws_score(ws, cap, c);
    const float* gbx = ws_box(ws, cap, c);
    const int*   gix = ws_idx(ws, cap, c);
#if USE_ASYNC_LDS
    for (int i = tid; i < cnt; i += 256) {
        __builtin_amdgcn_global_load_async_to_lds_b32(TO_GLB32(gsc + i),              TO_LDS32(&s_score[i]),   0, 0);
        __builtin_amdgcn_global_load_async_to_lds_b128(TO_GLB128(gbx + (size_t)i * 4), TO_LDS128(&s_box[i * 4]), 0, 0);
        __builtin_amdgcn_global_load_async_to_lds_b32(TO_GLB32(gix + i),              TO_LDS32(&s_idx[i]),     0, 0);
    }
# if __has_builtin(__builtin_amdgcn_s_wait_asynccnt)
    __builtin_amdgcn_s_wait_asynccnt(0);
# else
    asm volatile("s_wait_asynccnt 0" ::: "memory");
# endif
#else
    for (int i = tid; i < cnt; i += 256) {
        s_score[i] = gsc[i];
        s_box[i * 4 + 0] = gbx[i * 4 + 0];
        s_box[i * 4 + 1] = gbx[i * 4 + 1];
        s_box[i * 4 + 2] = gbx[i * 4 + 2];
        s_box[i * 4 + 3] = gbx[i * 4 + 3];
        s_idx[i] = gix[i];
    }
#endif
    __syncthreads();

    // greedy NMS, fixed NUM_DET iterations
    for (int it = 0; it < NUM_DET; ++it) {
        float bv = NEG;
        int   bi = -1;
        for (int i = tid; i < cnt; i += 256) {
            float s = s_score[i];
            if (s > bv) { bv = s; bi = i; }
        }
        red_val[tid] = bv;
        red_idx[tid] = bi;
        __syncthreads();
        for (int off = 128; off > 0; off >>= 1) {
            if (tid < off) {
                if (red_val[tid + off] > red_val[tid]) {
                    red_val[tid] = red_val[tid + off];
                    red_idx[tid] = red_idx[tid + off];
                }
            }
            __syncthreads();
        }
        if (tid == 0) {
            int   wbi = red_idx[0];
            float wbv = red_val[0];
            if (wbi >= 0 && wbv > NEG) {
                s_keep[it]   = wbi;
                s_kscore[it] = wbv;
                s_bb[0] = s_box[wbi * 4 + 0];
                s_bb[1] = s_box[wbi * 4 + 1];
                s_bb[2] = s_box[wbi * 4 + 2];
                s_bb[3] = s_box[wbi * 4 + 3];
                s_bi = wbi;
            } else {
                s_keep[it]   = -1;
                s_kscore[it] = 0.0f;
                s_bi = -1;
            }
        }
        __syncthreads();
        if (s_bi >= 0) {
            float bx1 = s_bb[0], by1 = s_bb[1], bx2 = s_bb[2], by2 = s_bb[3];
            float areaB = (bx2 - bx1) * (by2 - by1);
            for (int i = tid; i < cnt; i += 256) {
                float x1 = s_box[i * 4 + 0], y1 = s_box[i * 4 + 1];
                float x2 = s_box[i * 4 + 2], y2 = s_box[i * 4 + 3];
                float lx = fmaxf(bx1, x1), ly = fmaxf(by1, y1);
                float rx = fminf(bx2, x2), ry = fminf(by2, y2);
                float iw = fmaxf(rx - lx, 0.0f), ih = fmaxf(ry - ly, 0.0f);
                float inter = iw * ih;
                float areaA = (x2 - x1) * (y2 - y1);
                float iou = inter / (areaB + areaA - inter + 1e-9f);
                if (iou > NMS_TH) s_score[i] = NEG;   // suppress (incl. self)
            }
        }
        __syncthreads();
    }

    // outputs: boxes[0,36000) labels[36000,45000) scores[45000,54000)
    //          future[54000,162000) valid[162000,171000)  (4-byte elems)
    float* out_boxes  = out;
    int*   out_labels = (int*)(out + M_OUT * 4);
    float* out_scores = out + M_OUT * 4 + M_OUT;
    float* out_future = out + M_OUT * 4 + M_OUT * 2;
    float* out_valid  = out + M_OUT * 4 + M_OUT * 2 + (size_t)NT * M_OUT * 4;

    const float imw = (float)(*pW);
    const float imh = (float)(*pH);

    if (tid < NUM_DET) {
        const int k    = tid;
        const int m    = c * NUM_DET + k;
        const int kept = s_keep[k];
        out_labels[m] = c + 1;
        if (kept >= 0) {
            out_boxes[m * 4 + 0] = s_box[kept * 4 + 0];
            out_boxes[m * 4 + 1] = s_box[kept * 4 + 1];
            out_boxes[m * 4 + 2] = s_box[kept * 4 + 2];
            out_boxes[m * 4 + 3] = s_box[kept * 4 + 3];
            out_scores[m] = s_kscore[k];
            out_valid[m]  = 1.0f;

            // iterative trajectory decode starting from the raw proposal
            int row = s_idx[kept];
            float4 pr = ((const float4*)proposal)[row];
            float cx1 = pr.x, cy1 = pr.y, cx2 = pr.z, cy2 = pr.w;
#pragma unroll
            for (int t = 0; t < NT; ++t) {
                float4 dl = ((const float4*)traj)[((size_t)t * NPROP + row) * NCLS + (c + 1)];
                float nx1, ny1, nx2, ny2;
                decode1(dl.x, dl.y, dl.z, dl.w, cx1, cy1, cx2, cy2, nx1, ny1, nx2, ny2);
                float* of = out_future + ((size_t)t * M_OUT + m) * 4;
                of[0] = clampf(nx1, 0.0f, imw);
                of[1] = clampf(ny1, 0.0f, imh);
                of[2] = clampf(nx2, 0.0f, imw);
                of[3] = clampf(ny2, 0.0f, imh);
                cx1 = nx1; cy1 = ny1; cx2 = nx2; cy2 = ny2;  // chain uses unclipped
            }
        } else {
            out_boxes[m * 4 + 0] = 0.0f;
            out_boxes[m * 4 + 1] = 0.0f;
            out_boxes[m * 4 + 2] = 0.0f;
            out_boxes[m * 4 + 3] = 0.0f;
            out_scores[m] = 0.0f;
            out_valid[m]  = 0.0f;
#pragma unroll
            for (int t = 0; t < NT; ++t) {
                float* of = out_future + ((size_t)t * M_OUT + m) * 4;
                of[0] = 0.0f; of[1] = 0.0f; of[2] = 0.0f; of[3] = 0.0f;
            }
        }
    }
}

// ---------------------------------------------------------------------
extern "C" void kernel_launch(void* const* d_in, const int* in_sizes, int n_in,
                              void* d_out, int out_size, void* d_ws, size_t ws_size,
                              hipStream_t stream) {
    (void)in_sizes; (void)n_in; (void)out_size;
    const float* logits = (const float*)d_in[0];   // [N, 91]
    const float* boxreg = (const float*)d_in[1];   // [N, 364]
    const float* traj   = (const float*)d_in[2];   // [3, N, 364]
    const float* prop   = (const float*)d_in[3];   // [N, 4]
    const int*   pH     = (const int*)d_in[4];
    const int*   pW     = (const int*)d_in[5];
    unsigned char* ws   = (unsigned char*)d_ws;
    float* out          = (float*)d_out;

    int cap = CAP_MAX;
    size_t need = 512 + (size_t)NC * cap * 24;
    if (ws_size < need) {
        long avail = (long)ws_size - 512;
        cap = avail > 0 ? (int)(avail / ((long)NC * 24)) : 64;
        cap &= ~63;                 // keep 16B alignment of per-class sections
        if (cap < 64) cap = 64;
        if (cap > CAP_MAX) cap = CAP_MAX;
    }

    k_zero_counters<<<1, 128, 0, stream>>>((int*)ws);

    int blocksB = (NPROP + 127) / 128;   // 128 rows per block (8 waves x 16)
    k_softmax_filter<<<blocksB, 256, 0, stream>>>(logits, boxreg, prop, ws, cap, pH, pW);

    k_nms_output<<<NC, 256, 0, stream>>>(ws, cap, traj, prop, out, pH, pW);
}
